// CRNN_12841952215391
// MI455X (gfx1250) — matrix operand order repaired and checked by reading, not compiled, not run on hardware
//
#include <hip/hip_runtime.h>

// ---------------------------------------------------------------------------
// CRNN for MI455X (gfx1250): bf16 WMMA GEMMs everywhere GEMM-shaped work exists.
// Wave tile 32x64 (2 A-frags x 4 B-frags -> 8 v_wmma per K-step, 12 b128 loads),
// 4 waves stacked in M per block (block tile 128x64), __launch_bounds__(128,1)
// so the 8 f32 accumulators + 6 fragments live entirely in VGPRs (no spills).
// Conv layers: implicit-im2col WMMA GEMM, K ordered (dy,dx,cin) so each K-block
// of 32 is a contiguous cin run -> 2x b128 loads per fragment per lane.
// LSTM: only layer 1 matters (reference bug: hs_all[-1], layers un-stacked).
// x@Wi hoisted to one 2048x2048x2048 WMMA GEMM per direction; h@Wh per step.
// ---------------------------------------------------------------------------

typedef __attribute__((ext_vector_type(16))) __bf16 v16bf;
typedef __attribute__((ext_vector_type(8)))  float  v8f;

union FragBF {
  v16bf v;
  uint4 q[2];
  unsigned short e[16];
};

__device__ __forceinline__ unsigned short f2bf(float f) {
  unsigned u = __float_as_uint(f);
  u += 0x7fffu + ((u >> 16) & 1u);          // round-to-nearest-even
  return (unsigned short)(u >> 16);
}
__device__ __forceinline__ float bf2f(unsigned short h) {
  return __uint_as_float(((unsigned)h) << 16);
}
__device__ __forceinline__ v8f wmma_bf(v16bf a, v16bf b, v8f c) {
  return __builtin_amdgcn_wmma_f32_16x16x32_bf16(false, a, false, b,
                                                 (short)0, c, false, false);
}

// ---------------- elementwise helpers ----------------

__global__ void k_cvt(const float* __restrict__ in, unsigned short* __restrict__ out, int n) {
  int i = blockIdx.x * blockDim.x + threadIdx.x;
  if (i < n) out[i] = f2bf(in[i]);
}

__global__ void k_zero16(unsigned short* p, int n) {
  int i = blockIdx.x * blockDim.x + threadIdx.x;
  if (i < n) p[i] = 0;
}
__global__ void k_zero32(float* p, int n) {
  int i = blockIdx.x * blockDim.x + threadIdx.x;
  if (i < n) p[i] = 0.0f;
}

// in: fp32 [K,N] row-major   out: bf16 [N,Kpad] (i.e. B^T, zero-padded K)
__global__ void k_tc(const float* __restrict__ in, unsigned short* __restrict__ out,
                     int K, int N, int Kpad) {
  int idx = blockIdx.x * blockDim.x + threadIdx.x;
  if (idx >= N * Kpad) return;
  int n = idx / Kpad, k = idx - n * Kpad;
  out[idx] = (k < K) ? f2bf(in[(size_t)k * N + n]) : (unsigned short)0;
}

// ---------------- WMMA GEMM: C = A[M,K] * B (Bt = [N,K] col-major) ----------
// Block: 128 thr = 4 waves stacked in M; wave tile 32x64.
// Grid: x = N/64, y = ceil(M/128). Rows clamped on load (EXEC stays all-ones),
// stores guarded, so M need not be a multiple of 128.
// mode 0: none   mode 1: += bias[n] (fp32)   mode 2: += add[m*N+n] (fp32)
__global__ void __launch_bounds__(128, 1)
k_gemm(const unsigned short* __restrict__ A,
       const unsigned short* __restrict__ Bt,
       const float* __restrict__ add,
       float* __restrict__ outF,
       int M, int N, int K, int mode) {
  int lane = threadIdx.x & 31;
  int wave = threadIdx.x >> 5;
  int m0 = blockIdx.y * 128 + wave * 32;
  int n0 = blockIdx.x * 64;
  int half = lane >> 4, l16 = lane & 15;

  v8f acc[2][4] = {};
  for (int k0 = 0; k0 < K; k0 += 32) {
    FragBF a[2], b[4];
    for (int s = 0; s < 2; s++) {
      int row = m0 + s * 16 + l16;
      if (row >= M) row = M - 1;                        // branch-free clamp
      const unsigned short* pa = A + (size_t)row * K + k0 + half * 8;
      a[s].q[0] = *(const uint4*)pa;
      a[s].q[1] = *(const uint4*)(pa + 16);
    }
    // unconditional speculative prefetch of the streaming operand, 2 K-tiles
    // ahead (global_prefetch_b8; safe past the end, no LOADcnt, no branches)
    __builtin_prefetch(A + (size_t)(m0 + l16) * K + k0 + half * 8 + 64, 0, 1);
    for (int j = 0; j < 4; j++) {
      const unsigned short* pb = Bt + (size_t)(n0 + j * 16 + l16) * K + k0 + half * 8;
      b[j].q[0] = *(const uint4*)pb;
      b[j].q[1] = *(const uint4*)(pb + 16);
    }
    for (int i = 0; i < 2; i++)
      for (int j = 0; j < 4; j++)
        acc[i][j] = wmma_bf(a[i].v, b[j].v, acc[i][j]);
  }
  for (int i = 0; i < 2; i++)
    for (int j = 0; j < 4; j++)
      for (int r = 0; r < 8; r++) {
        int gm = m0 + i * 16 + r + half * 8;
        int gn = n0 + j * 16 + l16;
        if (gm < M) {
          float v = acc[i][j][r];
          if (mode == 1) v += add[gn];
          else if (mode == 2) v += add[(size_t)gm * N + gn];
          outF[(size_t)gm * N + gn] = v;
        }
      }
}

// ------------- implicit-im2col conv3x3 SAME + bias + ReLU + BN (eval) -------
// act: bf16 NHWC [NB,H,W,Cin]; Wt: bf16 [Cout,Kpad] (col-major of [9*Cin,Cout])
// out: bf16 NHWC [NB,H,W,Cout].  M = NB*H*W is a multiple of 128 for every
// layer here, and Cout a multiple of 64, so no guards needed.
__global__ void __launch_bounds__(128, 1)
k_conv(const unsigned short* __restrict__ act,
       const unsigned short* __restrict__ Wt,
       const float* __restrict__ bias, const float* __restrict__ bs,
       const float* __restrict__ bo, const float* __restrict__ bm,
       const float* __restrict__ bv,
       unsigned short* __restrict__ out,
       int NB, int H, int W, int Cin, int Cout, int Kpad) {
  int lane = threadIdx.x & 31;
  int wave = threadIdx.x >> 5;
  int m0 = blockIdx.y * 128 + wave * 32;
  int n0 = blockIdx.x * 64;
  int half = lane >> 4, l16 = lane & 15;
  (void)NB;

  int an[2], ay[2], ax[2];
  for (int s = 0; s < 2; s++) {
    int row = m0 + s * 16 + l16;
    int hw = H * W;
    an[s] = row / hw;
    int rem = row - an[s] * hw;
    ay[s] = rem / W;
    ax[s] = rem - ay[s] * W;
  }

  v8f acc[2][4] = {};
  for (int k0 = 0; k0 < Kpad; k0 += 32) {
    FragBF a[2], b[4];
    for (int j = 0; j < 4; j++) {
      const unsigned short* pb = Wt + (size_t)(n0 + j * 16 + l16) * Kpad + k0 + half * 8;
      b[j].q[0] = *(const uint4*)pb;
      b[j].q[1] = *(const uint4*)(pb + 16);
    }
    for (int s = 0; s < 2; s++) {
      if ((Cin & 31) == 0) {              // uniform branch: conv2..conv5
        int p = k0 / Cin, c0 = k0 - p * Cin;
        int yy = ay[s] + p / 3 - 1;
        int xx = ax[s] + p % 3 - 1;
        if (yy >= 0 && yy < H && xx >= 0 && xx < W) {
          const unsigned short* pa =
              act + (((size_t)an[s] * H + yy) * W + xx) * Cin + c0 + half * 8;
          a[s].q[0] = *(const uint4*)pa;
          a[s].q[1] = *(const uint4*)(pa + 16);
        } else {
          a[s].q[0] = make_uint4(0, 0, 0, 0);
          a[s].q[1] = make_uint4(0, 0, 0, 0);
        }
      } else {                            // conv1 (Cin==1, K=9 padded to 32)
        for (int i = 0; i < 16; i++) {
          int ke = k0 + (i >= 8 ? 16 : 0) + half * 8 + (i & 7);
          unsigned short vv = 0;
          if (ke < 9 * Cin) {
            int p = ke / Cin, ci = ke - p * Cin;
            int yy = ay[s] + p / 3 - 1;
            int xx = ax[s] + p % 3 - 1;
            if (yy >= 0 && yy < H && xx >= 0 && xx < W)
              vv = act[(((size_t)an[s] * H + yy) * W + xx) * Cin + ci];
          }
          a[s].e[i] = vv;
        }
      }
    }
    for (int i = 0; i < 2; i++)
      for (int j = 0; j < 4; j++)
        acc[i][j] = wmma_bf(a[i].v, b[j].v, acc[i][j]);
  }

  for (int i = 0; i < 2; i++)
    for (int j = 0; j < 4; j++)
      for (int r = 0; r < 8; r++) {
        int gm = m0 + i * 16 + r + half * 8;
        int gn = n0 + j * 16 + l16;
        float v = acc[i][j][r] + bias[gn];
        v = fmaxf(v, 0.0f);                                   // ReLU
        float sc = bs[gn] * rsqrtf(bv[gn] + 1e-5f);           // BN eval
        v = (v - bm[gn]) * sc + bo[gn];
        out[(size_t)gm * Cout + gn] = f2bf(v);
      }
}

// ---------------- 2x2 max pool, stride 2 (all dims even) --------------------
__global__ void k_pool(const unsigned short* __restrict__ in,
                       unsigned short* __restrict__ out,
                       int NB, int H, int W, int C) {
  int Ho = H >> 1, Wo = W >> 1;
  size_t total = (size_t)NB * Ho * Wo * C;
  size_t idx = (size_t)blockIdx.x * blockDim.x + threadIdx.x;
  if (idx >= total) return;
  int c = (int)(idx % C);
  size_t t = idx / C;
  int x = (int)(t % Wo); t /= Wo;
  int y = (int)(t % Ho);
  int n = (int)(t / Ho);
  const unsigned short* p = in + (((size_t)n * H + 2 * y) * W + 2 * x) * C + c;
  float v0 = bf2f(p[0]), v1 = bf2f(p[C]);
  float v2 = bf2f(p[(size_t)W * C]), v3 = bf2f(p[(size_t)W * C + C]);
  out[idx] = f2bf(fmaxf(fmaxf(v0, v1), fmaxf(v2, v3)));
}

// raw reshape + swapaxes: Aseq[(t*64+b)*2048+d] = conv5[b*65536 + tt*2048 + d]
__global__ void k_seq(const unsigned short* __restrict__ c5,
                      unsigned short* __restrict__ Aseq, int rev) {
  int idx = blockIdx.x * blockDim.x + threadIdx.x;
  if (idx >= 2048 * 2048) return;
  int row = idx >> 11, d = idx & 2047;
  int t = row >> 6, b = row & 63;
  int tt = rev ? 31 - t : t;
  Aseq[idx] = c5[(size_t)b * 65536 + (size_t)tt * 2048 + d];
}

// LSTM pointwise gates: z fp32 [64,2048] -> update c, h; record h into hs[B,T,512]
__global__ void k_gates(const float* __restrict__ z, float* __restrict__ c,
                        unsigned short* __restrict__ h,
                        unsigned short* __restrict__ hs, int t) {
  int idx = blockIdx.x * blockDim.x + threadIdx.x;
  if (idx >= 64 * 512) return;
  int b = idx >> 9, j = idx & 511;
  const float* zr = z + (size_t)b * 2048;
  float zi = zr[j], zf = zr[j + 512], zg = zr[j + 1024], zo = zr[j + 1536];
  float si = 1.0f / (1.0f + __expf(-zi));
  float sf = 1.0f / (1.0f + __expf(-zf));
  float so = 1.0f / (1.0f + __expf(-zo));
  float c2 = sf * c[idx] + si * tanhf(zg);
  float h2 = so * tanhf(c2);
  c[idx] = c2;
  h[idx] = f2bf(h2);
  hs[((size_t)b * 32 + t) * 512 + j] = f2bf(h2);
}

// FC (1024->80) + log_softmax, one block per (b,t)
__global__ void k_fc(const unsigned short* __restrict__ hsf,
                     const unsigned short* __restrict__ hsb,
                     const float* __restrict__ W, const float* __restrict__ bias,
                     float* __restrict__ out) {
  __shared__ float logits[80];
  __shared__ float lse;
  int bt = blockIdx.x;
  const unsigned short* xf = hsf + (size_t)bt * 512;
  const unsigned short* xb = hsb + (size_t)bt * 512;
  for (int j = threadIdx.x; j < 80; j += blockDim.x) {
    float acc = bias[j];
    for (int k = 0; k < 512; k++) acc += bf2f(xf[k]) * W[(size_t)k * 80 + j];
    for (int k = 0; k < 512; k++) acc += bf2f(xb[k]) * W[(size_t)(512 + k) * 80 + j];
    logits[j] = acc;
  }
  __syncthreads();
  if (threadIdx.x == 0) {
    float mx = -1e30f;
    for (int j = 0; j < 80; j++) mx = fmaxf(mx, logits[j]);
    float s = 0.0f;
    for (int j = 0; j < 80; j++) s += __expf(logits[j] - mx);
    lse = mx + __logf(s);
  }
  __syncthreads();
  float l = lse;
  for (int j = threadIdx.x; j < 80; j += blockDim.x)
    out[(size_t)bt * 80 + j] = logits[j] - l;
}

// ---------------------------------------------------------------------------

extern "C" void kernel_launch(void* const* d_in, const int* in_sizes, int n_in,
                              void* d_out, int out_size, void* d_ws, size_t ws_size,
                              hipStream_t stream) {
  (void)in_sizes; (void)n_in; (void)out_size; (void)ws_size;

  const float* x = (const float*)d_in[0];
  const float *Kw[5], *Bw[5], *Sw[5], *Ow[5], *Mw[5], *Vw[5];
  for (int i = 0; i < 5; i++) {
    Kw[i] = (const float*)d_in[1 + 6 * i];
    Bw[i] = (const float*)d_in[2 + 6 * i];
    Sw[i] = (const float*)d_in[3 + 6 * i];
    Ow[i] = (const float*)d_in[4 + 6 * i];
    Mw[i] = (const float*)d_in[5 + 6 * i];
    Vw[i] = (const float*)d_in[6 + 6 * i];
  }
  const float* fw_Wi = (const float*)d_in[31];
  const float* fw_Wh = (const float*)d_in[32];
  const float* fw_b  = (const float*)d_in[33];
  const float* bw_Wi = (const float*)d_in[34];
  const float* bw_Wh = (const float*)d_in[35];
  const float* bw_b  = (const float*)d_in[36];
  const float* fcW   = (const float*)d_in[37];
  const float* fcb   = (const float*)d_in[38];
  float* out = (float*)d_out;
  char* ws = (char*)d_ws;

  // workspace layout (arena reuse keeps total ~106 MB)
  size_t off = 0;
  auto al = [&](size_t b) { size_t o = off; off += (b + 255) & ~(size_t)255; return o; };
  size_t o_xb  = al(524288ull * 2);
  size_t o_wt1 = al(32ull * 64 * 2);
  size_t o_wt2 = al(576ull * 128 * 2);
  size_t o_wt3 = al(1152ull * 256 * 2);
  size_t o_wt4 = al(2304ull * 256 * 2);
  size_t o_wt5 = al(2304ull * 512 * 2);
  size_t o_ar  = al(67108864ull);   // conv1 out; later c2; later whole LSTM set
  size_t o_p1  = al(16777216ull);   // pool1; later c3; later c5
  size_t o_p2  = al(8388608ull);    // pool2; later c4
  size_t o_p3  = al(4194304ull);
  size_t o_hsf = al(2097152ull);
  size_t o_hsb = al(2097152ull);

  unsigned short* xb  = (unsigned short*)(ws + o_xb);
  unsigned short* wt1 = (unsigned short*)(ws + o_wt1);
  unsigned short* wt2 = (unsigned short*)(ws + o_wt2);
  unsigned short* wt3 = (unsigned short*)(ws + o_wt3);
  unsigned short* wt4 = (unsigned short*)(ws + o_wt4);
  unsigned short* wt5 = (unsigned short*)(ws + o_wt5);
  unsigned short* c1  = (unsigned short*)(ws + o_ar);
  unsigned short* p1  = (unsigned short*)(ws + o_p1);
  unsigned short* c2  = c1;                                   // reuse
  unsigned short* p2  = (unsigned short*)(ws + o_p2);
  unsigned short* c3  = p1;                                   // reuse
  unsigned short* p3  = (unsigned short*)(ws + o_p3);
  unsigned short* c4  = p2;                                   // reuse
  unsigned short* c5  = p1;                                   // reuse
  unsigned short* aseq = (unsigned short*)(ws + o_ar);
  unsigned short* wiT  = (unsigned short*)(ws + o_ar + 8388608ull);
  unsigned short* whT  = (unsigned short*)(ws + o_ar + 16777216ull);
  float* zx    = (float*)(ws + o_ar + 18874368ull);
  float* zbuf  = (float*)(ws + o_ar + 35651584ull);
  unsigned short* hbuf = (unsigned short*)(ws + o_ar + 36175872ull);
  float* cbuf  = (float*)(ws + o_ar + 36241408ull);
  unsigned short* hsf = (unsigned short*)(ws + o_hsf);
  unsigned short* hsb = (unsigned short*)(ws + o_hsb);

  // input + weight conversion
  k_cvt<<<2048, 256, 0, stream>>>(x, xb, 524288);
  k_tc<<<(64 * 32 + 255) / 256, 256, 0, stream>>>(Kw[0], wt1, 9, 64, 32);
  k_tc<<<(128 * 576 + 255) / 256, 256, 0, stream>>>(Kw[1], wt2, 576, 128, 576);
  k_tc<<<(256 * 1152 + 255) / 256, 256, 0, stream>>>(Kw[2], wt3, 1152, 256, 1152);
  k_tc<<<(256 * 2304 + 255) / 256, 256, 0, stream>>>(Kw[3], wt4, 2304, 256, 2304);
  k_tc<<<(512 * 2304 + 255) / 256, 256, 0, stream>>>(Kw[4], wt5, 2304, 512, 2304);

  // CNN stem (block tile 128x64; M = NB*H*W, grid = (Cout/64, M/128))
  k_conv<<<dim3(1, 4096), 128, 0, stream>>>(xb, wt1, Bw[0], Sw[0], Ow[0], Mw[0], Vw[0],
                                            c1, 64, 32, 256, 1, 64, 32);
  k_pool<<<(8388608 + 255) / 256, 256, 0, stream>>>(c1, p1, 64, 32, 256, 64);
  k_conv<<<dim3(2, 1024), 128, 0, stream>>>(p1, wt2, Bw[1], Sw[1], Ow[1], Mw[1], Vw[1],
                                            c2, 64, 16, 128, 64, 128, 576);
  k_pool<<<(4194304 + 255) / 256, 256, 0, stream>>>(c2, p2, 64, 16, 128, 128);
  k_conv<<<dim3(4, 256), 128, 0, stream>>>(p2, wt3, Bw[2], Sw[2], Ow[2], Mw[2], Vw[2],
                                           c3, 64, 8, 64, 128, 256, 1152);
  k_pool<<<(2097152 + 255) / 256, 256, 0, stream>>>(c3, p3, 64, 8, 64, 256);
  k_conv<<<dim3(4, 64), 128, 0, stream>>>(p3, wt4, Bw[3], Sw[3], Ow[3], Mw[3], Vw[3],
                                          c4, 64, 4, 32, 256, 256, 2304);
  k_conv<<<dim3(8, 64), 128, 0, stream>>>(c4, wt5, Bw[4], Sw[4], Ow[4], Mw[4], Vw[4],
                                          c5, 64, 4, 32, 256, 512, 2304);

  // BiLSTM (only layer 1 of each direction matters; see reference hs_all[-1])
  for (int dir = 0; dir < 2; dir++) {
    const float* Wi = dir ? bw_Wi : fw_Wi;
    const float* Wh = dir ? bw_Wh : fw_Wh;
    const float* bb = dir ? bw_b : fw_b;
    unsigned short* hs = dir ? hsb : hsf;

    k_seq<<<(2048 * 2048) / 256, 256, 0, stream>>>(c5, aseq, dir);
    k_tc<<<(2048 * 2048) / 256, 256, 0, stream>>>(Wi + 4194304, wiT, 2048, 2048, 2048);
    k_tc<<<(2048 * 512) / 256, 256, 0, stream>>>(Wh + 1048576, whT, 512, 2048, 512);
    // Zx = X @ Wi1 + b1  :  [2048,2048] x [2048,2048]
    k_gemm<<<dim3(32, 16), 128, 0, stream>>>(aseq, wiT, bb + 2048, zx,
                                             2048, 2048, 2048, 1);
    k_zero16<<<128, 256, 0, stream>>>(hbuf, 32768);
    k_zero32<<<128, 256, 0, stream>>>(cbuf, 32768);
    for (int t = 0; t < 32; t++) {
      // z = Zx[t] + h @ Wh1  :  [64,2048] += [64,512] x [512,2048]
      k_gemm<<<dim3(32, 1), 128, 0, stream>>>(hbuf, whT, zx + (size_t)t * 64 * 2048,
                                              zbuf, 64, 2048, 512, 2);
      k_gates<<<128, 256, 0, stream>>>(zbuf, cbuf, hbuf, hs, t);
    }
  }

  // FC + log_softmax
  k_fc<<<2048, 128, 0, stream>>>(hsf, hsb, fcW, fcb, out);
}
